// DeformableTransformer_14937896256230
// MI455X (gfx1250) — matrix operand order repaired and checked by reading, not compiled, not run on hardware
//
#include <hip/hip_runtime.h>
#include <hip/hip_bf16.h>

// ---------------- problem constants (match reference) ----------------
#define D3v   8
#define H3v   32
#define W3v   32
#define NTOK  8192          // D3*H3*W3
#define DMODEL 256
#define NHEADS 8
#define CHAN   32           // DMODEL / NHEADS
#define NPTS   4
#define DFFN   1024
#define NLAY   6
#define NQv    100
#define BSv    2
#define EPSv   1e-5f

typedef __attribute__((ext_vector_type(16))) __bf16 v16bf;
typedef __attribute__((ext_vector_type(8)))  __bf16 v8bf;
typedef __attribute__((ext_vector_type(8)))  float  v8f;

// ==================================================================
// GEMM: C[M,N] = A[M,K] @ W[N,K]^T + bias, optional ReLU, f32 and/or
// bf16 output. A and W are bf16 (pre-materialized). Staging uses
// GLOBAL_LOAD_ASYNC_TO_LDS_B128 (ASYNCcnt) with double-buffered LDS:
// per wave per K-step only 4 async DMA instructions, no VGPR staging,
// no conversions in the hot loop. Block tile 128x128, 8 waves, each
// wave owns 32x64 (2x4 of 16x16 v_wmma_f32_16x16x32_bf16).
// ==================================================================
#define BM 128
#define BN 128
#define BK 32
#define LDSTR 40                 // bf16 elems per LDS row (80B, 16B-aligned)
#define TILE_E (256 * LDSTR)     // elems per buffer: A rows 0..127, B rows 128..255
#define TILE_BYTES (TILE_E * 2)  // 20480

__global__ __launch_bounds__(256)
void gemm_bf16_kernel(const __bf16* __restrict__ A, const __bf16* __restrict__ W,
                      const float* __restrict__ bias,
                      float* __restrict__ Cf, __bf16* __restrict__ Cbf,
                      int M, int N, int K, int relu)
{
    __shared__ __align__(16) __bf16 smem[2 * TILE_E];

    const int tid  = threadIdx.x;
    const int wv   = tid >> 5;
    const int lane = tid & 31;
    const int wm   = wv >> 1;        // 0..3
    const int wn   = wv & 1;         // 0..1
    const int bm0  = blockIdx.y * BM;
    const int bn0  = blockIdx.x * BN;
    const int hl   = lane >> 4;
    const int r    = lane & 15;

    // Async staging: each wave copies 16 A rows + 16 B rows per K-step,
    // 2 b128 instructions each (8 rows per instr: 4 lanes x 16B per row).
    const int la = lane >> 2;            // row within group of 8
    const int lc = (lane & 3) * 8;       // col chunk (8 bf16 = 16B)
    const int r0 = wv * 16 + la;         // local tile rows
    const int r1 = r0 + 8;

    auto gclamp = [](int v, int hi) { return v < hi ? v : hi - 1; };
    // OOB rows clamped to last valid row: duplicate reads, masked at store.
    const unsigned long long ga0 = (unsigned long long)(A + (size_t)gclamp(bm0 + r0, M) * K + lc);
    const unsigned long long ga1 = (unsigned long long)(A + (size_t)gclamp(bm0 + r1, M) * K + lc);
    const unsigned long long gb0 = (unsigned long long)(W + (size_t)gclamp(bn0 + r0, N) * K + lc);
    const unsigned long long gb1 = (unsigned long long)(W + (size_t)gclamp(bn0 + r1, N) * K + lc);

    const unsigned lA0 = (unsigned)(uintptr_t)&smem[(size_t)r0 * LDSTR + lc];
    const unsigned lA1 = (unsigned)(uintptr_t)&smem[(size_t)r1 * LDSTR + lc];
    const unsigned lB0 = (unsigned)(uintptr_t)&smem[(size_t)(128 + r0) * LDSTR + lc];
    const unsigned lB1 = (unsigned)(uintptr_t)&smem[(size_t)(128 + r1) * LDSTR + lc];

    auto issue = [&](int buf, int k0) {
        const unsigned bo = buf ? (unsigned)TILE_BYTES : 0u;
        const unsigned long long kb = (unsigned long long)k0 * 2;
        asm volatile("global_load_async_to_lds_b128 %0, %1, off"
                     :: "v"(lA0 + bo), "v"(ga0 + kb) : "memory");
        asm volatile("global_load_async_to_lds_b128 %0, %1, off"
                     :: "v"(lA1 + bo), "v"(ga1 + kb) : "memory");
        asm volatile("global_load_async_to_lds_b128 %0, %1, off"
                     :: "v"(lB0 + bo), "v"(gb0 + kb) : "memory");
        asm volatile("global_load_async_to_lds_b128 %0, %1, off"
                     :: "v"(lB1 + bo), "v"(gb1 + kb) : "memory");
    };

    v8f acc[2][4];
#pragma unroll
    for (int i = 0; i < 2; ++i)
#pragma unroll
        for (int j = 0; j < 4; ++j) acc[i][j] = (v8f)(0.0f);

    issue(0, 0);                       // prologue DMA into buffer 0
    int buf = 0;
    for (int k0 = 0; k0 < K; k0 += BK) {
        if (k0 + BK < K) {
            issue(buf ^ 1, k0 + BK);   // overlap next tile DMA with WMMA
            asm volatile("s_wait_asynccnt 4" ::: "memory");  // current tile done
        } else {
            asm volatile("s_wait_asynccnt 0" ::: "memory");
        }
        __syncthreads();               // all waves' DMA for this buffer landed

        const __bf16* tb = &smem[buf ? TILE_E : 0];
        // A fragments (ISA 16-bit A 16x32): lanes 0-15 hold K 0..7 & 16..23,
        // lanes 16-31 hold K 8..15 & 24..31; row = lane&15.
        union { v16bf v; v8bf h[2]; } afr[2];
#pragma unroll
        for (int sm = 0; sm < 2; ++sm) {
            const __bf16* ab = &tb[(wm * 32 + sm * 16 + r) * LDSTR + hl * 8];
            afr[sm].h[0] = *(const v8bf*)(ab);
            afr[sm].h[1] = *(const v8bf*)(ab + 16);
        }
        // B fragments (ISA 16-bit B 32x16): lanes 0-15 hold K 0..7 & 16..23,
        // lanes 16-31 hold K 8..15 & 24..31; col = lane&15.
        union { v16bf v; v8bf h[2]; } bfr[4];
#pragma unroll
        for (int sn = 0; sn < 4; ++sn) {
            const __bf16* bb = &tb[(128 + wn * 64 + sn * 16 + r) * LDSTR + hl * 8];
            bfr[sn].h[0] = *(const v8bf*)(bb);
            bfr[sn].h[1] = *(const v8bf*)(bb + 16);
        }
#pragma unroll
        for (int sn = 0; sn < 4; ++sn)
#pragma unroll
            for (int sm = 0; sm < 2; ++sm)
                acc[sm][sn] = __builtin_amdgcn_wmma_f32_16x16x32_bf16(
                    false, afr[sm].v, false, bfr[sn].v, (short)0, acc[sm][sn], false, false);
        __syncthreads();               // reads done before buffer is re-DMAed
        buf ^= 1;
    }

    // ---- epilogue: C/D layout — col = lane&15; VGPR i holds M = i + 8*(lane>=16)
#pragma unroll
    for (int sn = 0; sn < 4; ++sn) {
        int col = bn0 + wn * 64 + sn * 16 + r;
        if (col >= N) continue;
        float bv = bias ? bias[col] : 0.0f;
#pragma unroll
        for (int sm = 0; sm < 2; ++sm) {
#pragma unroll
            for (int i = 0; i < 8; ++i) {
                int row = bm0 + wm * 32 + sm * 16 + hl * 8 + i;
                if (row < M) {
                    float x = acc[sm][sn][i] + bv;
                    if (relu) x = fmaxf(x, 0.0f);
                    if (Cf)  Cf[(size_t)row * N + col]  = x;
                    if (Cbf) Cbf[(size_t)row * N + col] = (__bf16)x;
                }
            }
        }
    }
}

// ==================================================================
// f32 -> bf16 conversion (weights / initial activations)
// ==================================================================
__global__ void cvt_bf16_kernel(const float* __restrict__ in, __bf16* __restrict__ out, int n)
{
    int i = blockIdx.x * blockDim.x + threadIdx.x;
    if (i < n) out[i] = (__bf16)in[i];
}

// ==================================================================
// o_bf16 = a + b  (query = x + pos; feeds GEMMs only)
// ==================================================================
__global__ void add_bf_kernel(const float* __restrict__ a, const float* __restrict__ b,
                              __bf16* __restrict__ o, int n)
{
    int i = blockIdx.x * blockDim.x + threadIdx.x;
    if (i < n) o[i] = (__bf16)(a[i] + b[i]);
}

// ==================================================================
// Zero rows where mask is set (mask all-false in reference, applied anyway)
// ==================================================================
__global__ void mask_kernel(float* __restrict__ v, const unsigned char* __restrict__ mask, int n)
{
    int i = blockIdx.x * blockDim.x + threadIdx.x;
    if (i < n && mask[i >> 8]) v[i] = 0.0f;
}

// ==================================================================
// LayerNorm(x + res) * g + b -> f32 out + bf16 out (rows of 256)
// one wave32 per row; lane owns 8 contiguous elements
// ==================================================================
__global__ __launch_bounds__(256)
void resln_kernel(const float* __restrict__ x, const float* __restrict__ res,
                  const float* __restrict__ g, const float* __restrict__ b,
                  float* __restrict__ o, __bf16* __restrict__ obf, int rows)
{
    int row  = blockIdx.x * 8 + (threadIdx.x >> 5);
    int lane = threadIdx.x & 31;
    if (row >= rows) return;
    const float* xp = x   + (size_t)row * DMODEL + lane * 8;
    const float* rp = res + (size_t)row * DMODEL + lane * 8;
    float v[8];
    float s = 0.0f;
#pragma unroll
    for (int i = 0; i < 8; ++i) { v[i] = xp[i] + rp[i]; s += v[i]; }
#pragma unroll
    for (int off = 16; off > 0; off >>= 1) s += __shfl_xor(s, off, 32);
    float mean = s * (1.0f / DMODEL);
    float vs = 0.0f;
#pragma unroll
    for (int i = 0; i < 8; ++i) { float d = v[i] - mean; vs += d * d; }
#pragma unroll
    for (int off = 16; off > 0; off >>= 1) vs += __shfl_xor(vs, off, 32);
    float inv = rsqrtf(vs * (1.0f / DMODEL) + EPSv);
#pragma unroll
    for (int i = 0; i < 8; ++i) {
        int c = lane * 8 + i;
        float y = (v[i] - mean) * inv * g[c] + b[c];
        o[(size_t)row * DMODEL + c] = y;
        if (obf) obf[(size_t)row * DMODEL + c] = (__bf16)y;
    }
}

// ==================================================================
// Deformable attention sampling: one wave per (b, q, head), lane = channel.
// Softmax over NPTS logits + 8-corner trilinear gather; bf16 output
// (feeds the output-projection GEMM only).
// ==================================================================
__global__ __launch_bounds__(256)
void deform_sample_kernel(const float* __restrict__ v,    // (bs, NTOK, 256)
                          const float* __restrict__ off,  // (bs, Lq, 96)
                          const float* __restrict__ awl,  // (bs, Lq, 32) logits
                          const float* __restrict__ ref,  // (bs, Lq, 3)
                          __bf16* __restrict__ out,       // (bs, Lq, 256)
                          int Lq)
{
    int gw   = blockIdx.x * 8 + (threadIdx.x >> 5);
    int lane = threadIdx.x & 31;
    int total = BSv * Lq * NHEADS;
    if (gw >= total) return;
    int h = gw % NHEADS;
    int q = (gw / NHEADS) % Lq;
    int b = gw / (NHEADS * Lq);

    const float* refp = ref + ((size_t)b * Lq + q) * 3;
    float rx = refp[0], ry = refp[1], rz = refp[2];
    const float* offp = off + ((size_t)b * Lq + q) * (NHEADS * NPTS * 3) + h * NPTS * 3;
    const float* awp  = awl + ((size_t)b * Lq + q) * (NHEADS * NPTS) + h * NPTS;

    float a0 = awp[0], a1 = awp[1], a2 = awp[2], a3 = awp[3];
    float mx = fmaxf(fmaxf(a0, a1), fmaxf(a2, a3));
    float e0 = __expf(a0 - mx), e1 = __expf(a1 - mx), e2 = __expf(a2 - mx), e3 = __expf(a3 - mx);
    float inv = 1.0f / (e0 + e1 + e2 + e3);
    float aw[4] = { e0 * inv, e1 * inv, e2 * inv, e3 * inv };

    const float* vb = v + (size_t)b * NTOK * DMODEL + h * CHAN + lane;
    float accum = 0.0f;
#pragma unroll
    for (int p = 0; p < NPTS; ++p) {
        float lx = rx + offp[p * 3 + 0] * (1.0f / W3v);
        float ly = ry + offp[p * 3 + 1] * (1.0f / H3v);
        float lz = rz + offp[p * 3 + 2] * (1.0f / D3v);
        float gx = lx * W3v - 0.5f, gy = ly * H3v - 0.5f, gz = lz * D3v - 0.5f;
        float fx0 = floorf(gx), fy0 = floorf(gy), fz0 = floorf(gz);
        int x0 = (int)fx0, y0 = (int)fy0, z0 = (int)fz0;
        float fx = gx - fx0, fy = gy - fy0, fz = gz - fz0;
        float s = 0.0f;
#pragma unroll
        for (int dz = 0; dz < 2; ++dz)
#pragma unroll
            for (int dy = 0; dy < 2; ++dy)
#pragma unroll
                for (int dx = 0; dx < 2; ++dx) {
                    int xi = min(max(x0 + dx, 0), W3v - 1);
                    int yi = min(max(y0 + dy, 0), H3v - 1);
                    int zi = min(max(z0 + dz, 0), D3v - 1);
                    float w = (dx ? fx : 1.0f - fx) * (dy ? fy : 1.0f - fy) * (dz ? fz : 1.0f - fz);
                    int idx = (zi * H3v + yi) * W3v + xi;
                    s += w * vb[(size_t)idx * DMODEL];
                }
        accum += aw[p] * s;
    }
    out[((size_t)b * Lq + q) * DMODEL + h * CHAN + lane] = (__bf16)accum;
}

// ==================================================================
// Decoder self-attention: one block per (b, h, q). NQ=100, ch=32 — tiny.
// bf16 output (feeds sa_out GEMM only).
// ==================================================================
__global__ __launch_bounds__(128)
void mha_kernel(const float* __restrict__ Q, const float* __restrict__ Kx,
                const float* __restrict__ V, __bf16* __restrict__ O)
{
    int blk = blockIdx.x;
    int qi = blk % NQv;
    int h  = (blk / NQv) % NHEADS;
    int b  = blk / (NQv * NHEADS);
    __shared__ float s[NQv];
    __shared__ float red[2];

    const float* qp = Q + ((size_t)b * NQv + qi) * DMODEL + h * CHAN;
    int t = threadIdx.x;
    if (t < NQv) {
        const float* kp = Kx + ((size_t)b * NQv + t) * DMODEL + h * CHAN;
        float d = 0.0f;
#pragma unroll
        for (int c = 0; c < CHAN; ++c) d += qp[c] * kp[c];
        s[t] = d * 0.17677669529663689f;   // 1/sqrt(32)
    }
    __syncthreads();
    if (t == 0) {
        float m = -3.4e38f;
        for (int j = 0; j < NQv; ++j) m = fmaxf(m, s[j]);
        float ss = 0.0f;
        for (int j = 0; j < NQv; ++j) ss += __expf(s[j] - m);
        red[0] = m; red[1] = 1.0f / ss;
    }
    __syncthreads();
    if (t < NQv) s[t] = __expf(s[t] - red[0]) * red[1];
    __syncthreads();
    if (t < CHAN) {
        float o = 0.0f;
        for (int j = 0; j < NQv; ++j)
            o += s[j] * V[((size_t)b * NQv + j) * DMODEL + h * CHAN + t];
        O[((size_t)b * NQv + qi) * DMODEL + h * CHAN + t] = (__bf16)o;
    }
}

// ==================================================================
// ref = sigmoid(qe @ ref_w^T + ref_b)  — (bs, NQ, 3)
// ==================================================================
__global__ void ref_kernel(const float* __restrict__ qe_w, const float* __restrict__ rw,
                           const float* __restrict__ rb, float* __restrict__ ref)
{
    int i = blockIdx.x * blockDim.x + threadIdx.x;
    if (i >= BSv * NQv * 3) return;
    int d = i % 3;
    int q = (i / 3) % NQv;
    const float* x = qe_w + (size_t)q * (2 * DMODEL);   // first half
    float s = rb[d];
    for (int c = 0; c < DMODEL; ++c) s += x[c] * rw[d * DMODEL + c];
    ref[i] = 1.0f / (1.0f + __expf(-s));
}

// ==================================================================
// Broadcast query embedding halves: qe (f32), tgt (f32 + bf16)
// ==================================================================
__global__ void bcast_qe_kernel(const float* __restrict__ qe_w,
                                float* __restrict__ qe, float* __restrict__ tgt,
                                __bf16* __restrict__ tgt_bf)
{
    int i = blockIdx.x * blockDim.x + threadIdx.x;
    if (i >= BSv * NQv * DMODEL) return;
    int c = i % DMODEL;
    int q = (i / DMODEL) % NQv;
    float tv = qe_w[(size_t)q * (2 * DMODEL) + DMODEL + c];
    qe[i]     = qe_w[(size_t)q * (2 * DMODEL) + c];
    tgt[i]    = tv;
    tgt_bf[i] = (__bf16)tv;
}

// ==================================================================
// Host orchestration
// ==================================================================
enum {
    IN_SRCS = 0, IN_POINTS, IN_POS, IN_MASKS,
    E_OFF_W, E_OFF_B, E_AW_W, E_AW_B, E_VAL_W, E_VAL_B, E_OUT_W, E_OUT_B,
    E_LN1G, E_LN1B, E_LIN1W, E_LIN1B, E_LIN2W, E_LIN2B, E_LN2G, E_LN2B,
    D_OFF_W, D_OFF_B, D_AW_W, D_AW_B, D_VAL_W, D_VAL_B, D_OUT_W, D_OUT_B,
    D_LN1G, D_LN1B, D_LIN1W, D_LIN1B, D_LIN2W, D_LIN2B, D_LN2G, D_LN2B,
    D_SAINW, D_SAINB, D_SAOUTW, D_SAOUTB, D_LN3G, D_LN3B,
    IN_QE, IN_REFW, IN_REFB, IN_COUNT
};

extern "C" void kernel_launch(void* const* d_in, const int* in_sizes, int n_in,
                              void* d_out, int out_size, void* d_ws, size_t ws_size,
                              hipStream_t stream)
{
    const float* srcs   = (const float*)d_in[IN_SRCS];
    const float* points = (const float*)d_in[IN_POINTS];
    const float* pos    = (const float*)d_in[IN_POS];
    const unsigned char* masks = (const unsigned char*)d_in[IN_MASKS];
    const float* qe_w   = (const float*)d_in[IN_QE];
    const float* ref_w  = (const float*)d_in[IN_REFW];
    const float* ref_b  = (const float*)d_in[IN_REFB];

    auto LP = [&](int idx, int l) -> const float* {        // f32 per-layer param
        return (const float*)d_in[idx] + (size_t)(in_sizes[idx] / NLAY) * l;
    };

    // ---- workspace carve-out ----
    float* w = (float*)d_ws;
    size_t o = 0;
    auto alloc   = [&](size_t nf) { float* p = w + o; o += nf; return p; };
    auto allocbf = [&](size_t ne) { __bf16* p = (__bf16*)(w + o); o += (ne + 1) / 2; return p; };

    const size_t ACT  = (size_t)BSv * NTOK * DMODEL;   // 4.19M
    const size_t DACT = (size_t)BSv * NQv * DMODEL;    // 51200
    // f32 buffers
    float* mem  = alloc(ACT);
    float* vb   = alloc(ACT);
    float* proj = alloc(ACT);
    float* offb = alloc((size_t)BSv * NTOK * NHEADS * NPTS * 3);
    float* awb  = alloc((size_t)BSv * NTOK * NHEADS * NPTS);
    float* qe   = alloc(DACT);
    float* tgt  = alloc(DACT);
    float* refq = alloc((size_t)BSv * NQv * 3);
    float* dq   = alloc(DACT);
    float* dk   = alloc(DACT);
    float* dv   = alloc(DACT);
    float* dtmp = alloc(DACT);
    float* doff = alloc((size_t)BSv * NQv * NHEADS * NPTS * 3);
    float* daw  = alloc((size_t)BSv * NQv * NHEADS * NPTS);
    // bf16 activation buffers (GEMM inputs)
    __bf16* mem_bf  = allocbf(ACT);
    __bf16* qb_bf   = allocbf(ACT);
    __bf16* samp_bf = allocbf(ACT);
    __bf16* hb_bf   = allocbf((size_t)BSv * NTOK * DFFN);
    __bf16* tgt_bf  = allocbf(DACT);
    __bf16* dqin_bf = allocbf(DACT);
    __bf16* dmha_bf = allocbf(DACT);
    __bf16* dh_bf   = allocbf((size_t)BSv * NQv * DFFN);
    // bf16 weight pools (converted once per launch)
    __bf16* WB[IN_COUNT] = {};
    const int widx[14] = { E_VAL_W, E_OFF_W, E_AW_W, E_OUT_W, E_LIN1W, E_LIN2W,
                           D_VAL_W, D_OFF_W, D_AW_W, D_OUT_W, D_LIN1W, D_LIN2W,
                           D_SAINW, D_SAOUTW };
    for (int t = 0; t < 14; ++t) WB[widx[t]] = allocbf((size_t)in_sizes[widx[t]]);
    (void)ws_size; (void)n_in; (void)out_size;

    auto LPB = [&](int idx, int l) -> const __bf16* {      // bf16 per-layer weight
        return WB[idx] + (size_t)(in_sizes[idx] / NLAY) * l;
    };
    auto cvt = [&](const float* in, __bf16* out, int n) {
        cvt_bf16_kernel<<<(n + 255) / 256, 256, 0, stream>>>(in, out, n);
    };
    auto gemm = [&](const __bf16* A, const __bf16* W_, const float* bias,
                    float* Cf, __bf16* Cbf, int M, int N, int K, int relu) {
        dim3 g((N + BN - 1) / BN, (M + BM - 1) / BM);
        gemm_bf16_kernel<<<g, 256, 0, stream>>>(A, W_, bias, Cf, Cbf, M, N, K, relu);
    };
    auto addbf = [&](const float* a, const float* b, __bf16* c, int n) {
        add_bf_kernel<<<(n + 255) / 256, 256, 0, stream>>>(a, b, c, n);
    };
    auto resln = [&](const float* x, const float* r, const float* g_, const float* b_,
                     float* out_, __bf16* outbf_, int rows) {
        resln_kernel<<<(rows + 7) / 8, 256, 0, stream>>>(x, r, g_, b_, out_, outbf_, rows);
    };

    const int Me = BSv * NTOK;   // 16384 rows
    const int Md = BSv * NQv;    // 200 rows

    // one-time weight conversions (L2-resident, ~20 MB total)
    for (int t = 0; t < 14; ++t)
        cvt((const float*)d_in[widx[t]], WB[widx[t]], in_sizes[widx[t]]);

    // memory = srcs (f32 + bf16)
    hipMemcpyAsync(mem, srcs, ACT * sizeof(float), hipMemcpyDeviceToDevice, stream);
    cvt(srcs, mem_bf, (int)ACT);

    // ---------------- encoder ----------------
    for (int l = 0; l < NLAY; ++l) {
        addbf(mem, pos, qb_bf, (int)ACT);                                 // q = src + pos
        gemm(mem_bf, LPB(E_VAL_W, l), LP(E_VAL_B, l), vb, nullptr, Me, DMODEL, DMODEL, 0);
        mask_kernel<<<((int)ACT + 255) / 256, 256, 0, stream>>>(vb, masks, (int)ACT);
        gemm(qb_bf, LPB(E_OFF_W, l), LP(E_OFF_B, l), offb, nullptr, Me, NHEADS * NPTS * 3, DMODEL, 0);
        gemm(qb_bf, LPB(E_AW_W, l), LP(E_AW_B, l), awb, nullptr, Me, NHEADS * NPTS, DMODEL, 0);
        deform_sample_kernel<<<(BSv * NTOK * NHEADS + 7) / 8, 256, 0, stream>>>(
            vb, offb, awb, points, samp_bf, NTOK);
        gemm(samp_bf, LPB(E_OUT_W, l), LP(E_OUT_B, l), proj, nullptr, Me, DMODEL, DMODEL, 0);
        resln(proj, mem, LP(E_LN1G, l), LP(E_LN1B, l), mem, mem_bf, Me);
        gemm(mem_bf, LPB(E_LIN1W, l), LP(E_LIN1B, l), nullptr, hb_bf, Me, DFFN, DMODEL, 1);
        gemm(hb_bf, LPB(E_LIN2W, l), LP(E_LIN2B, l), proj, nullptr, Me, DMODEL, DFFN, 0);
        resln(proj, mem, LP(E_LN2G, l), LP(E_LN2B, l), mem, mem_bf, Me);
    }

    // ---------------- decoder prep ----------------
    bcast_qe_kernel<<<(BSv * NQv * DMODEL + 255) / 256, 256, 0, stream>>>(qe_w, qe, tgt, tgt_bf);
    ref_kernel<<<(BSv * NQv * 3 + 127) / 128, 128, 0, stream>>>(qe_w, ref_w, ref_b, refq);

    // ---------------- decoder ----------------
    for (int l = 0; l < NLAY; ++l) {
        const float* sa_b = LP(D_SAINB, l);                 // (768,)
        const __bf16* sa_wbf = LPB(D_SAINW, l);             // (768, 256) bf16
        addbf(tgt, qe, dqin_bf, (int)DACT);                 // q = tgt + qpos
        gemm(dqin_bf, sa_wbf,             sa_b,       dq, nullptr, Md, DMODEL, DMODEL, 0);
        gemm(dqin_bf, sa_wbf + 256 * 256, sa_b + 256, dk, nullptr, Md, DMODEL, DMODEL, 0);
        gemm(tgt_bf,  sa_wbf + 512 * 256, sa_b + 512, dv, nullptr, Md, DMODEL, DMODEL, 0);
        mha_kernel<<<BSv * NHEADS * NQv, 128, 0, stream>>>(dq, dk, dv, dmha_bf);
        gemm(dmha_bf, LPB(D_SAOUTW, l), LP(D_SAOUTB, l), dtmp, nullptr, Md, DMODEL, DMODEL, 0);
        resln(dtmp, tgt, LP(D_LN2G, l), LP(D_LN2B, l), tgt, tgt_bf, Md);

        // cross (deformable) attention against encoder memory
        addbf(tgt, qe, dqin_bf, (int)DACT);                 // q2 = tgt + qpos
        gemm(mem_bf, LPB(D_VAL_W, l), LP(D_VAL_B, l), vb, nullptr, Me, DMODEL, DMODEL, 0);
        mask_kernel<<<((int)ACT + 255) / 256, 256, 0, stream>>>(vb, masks, (int)ACT);
        gemm(dqin_bf, LPB(D_OFF_W, l), LP(D_OFF_B, l), doff, nullptr, Md, NHEADS * NPTS * 3, DMODEL, 0);
        gemm(dqin_bf, LPB(D_AW_W, l), LP(D_AW_B, l), daw, nullptr, Md, NHEADS * NPTS, DMODEL, 0);
        deform_sample_kernel<<<(BSv * NQv * NHEADS + 7) / 8, 256, 0, stream>>>(
            vb, doff, daw, refq, samp_bf, NQv);
        gemm(samp_bf, LPB(D_OUT_W, l), LP(D_OUT_B, l), dtmp, nullptr, Md, DMODEL, DMODEL, 0);
        resln(dtmp, tgt, LP(D_LN1G, l), LP(D_LN1B, l), tgt, tgt_bf, Md);

        // FFN
        gemm(tgt_bf, LPB(D_LIN1W, l), LP(D_LIN1B, l), nullptr, dh_bf, Md, DFFN, DMODEL, 1);
        gemm(dh_bf, LPB(D_LIN2W, l), LP(D_LIN2B, l), dtmp, nullptr, Md, DMODEL, DFFN, 0);
        resln(dtmp, tgt, LP(D_LN3G, l), LP(D_LN3B, l), tgt, tgt_bf, Md);
    }

    hipMemcpyAsync(d_out, tgt, DACT * sizeof(float), hipMemcpyDeviceToDevice, stream);
}
